// PPONetwork_4647154614942
// MI455X (gfx1250) — compile-verified, matrix operand order checked by reading, not run on hardware
//
#include <hip/hip_runtime.h>

// ---------------------------------------------------------------------------
// MI455X (gfx1250) fused MLP inference for the PPO card network.
// All GEMMs run on v_wmma_f32_16x16x32_bf16 (wave32). Weights are pre-packed
// into the ISA B-fragment layout (CDNA5 ISA 7.12.2) so the hot loop is pure
// global_load_b128 + v_wmma with 32-bit strength-reduced address induction.
// LayerNorm/ReLU/bias are fused epilogues; branch routing is done with
// stage-masked stores into one "fused" buffer.
// ---------------------------------------------------------------------------

typedef __bf16 bf16_t;
typedef __attribute__((ext_vector_type(16))) __bf16 v16bf;
typedef __attribute__((ext_vector_type(8)))  __bf16 v8bf;
typedef __attribute__((ext_vector_type(8)))  float  v8f;

__device__ __forceinline__ v16bf combine16(v8bf lo, v8bf hi) {
  v16bf r;
#pragma unroll
  for (int i = 0; i < 8; ++i) { r[i] = lo[i]; r[i + 8] = hi[i]; }
  return r;
}

// ---- generic WMMA GEMM + fused epilogue ------------------------------------
// 256 threads (8 wave32). Wave (wm,wn) computes MT m-tiles x NTW n-tiles.
// Split-K concat: k-tiles < ksplit read A1, rest read A2 (tile-aligned).
// Epilogue: bias (+LayerNorm) (+ReLU), bf16 or f32 out, optional stage mask.
template<int NTILES, int WAVES_N, int MT, bool LN, bool RELU, bool OUTF32, bool MASKED>
__global__ __launch_bounds__(256)
void gemm_kernel(const bf16_t* __restrict__ A1, int lda1,
                 const bf16_t* __restrict__ A2, int lda2, int ksplit,
                 int KT, const bf16_t* __restrict__ Wp,
                 const float* __restrict__ bias,
                 const float* __restrict__ gamma, const float* __restrict__ beta,
                 void* __restrict__ outp, int ldo,
                 const int* __restrict__ stage, int stageVal)
{
  constexpr int WAVES_M  = 8 / WAVES_N;
  constexpr int NTW      = NTILES / WAVES_N;
  constexpr int N        = NTILES * 16;
  constexpr int TOTAL_MT = WAVES_M * MT;
  constexpr int MBLOCK   = TOTAL_MT * 16;
  constexpr int LDSS     = N + 4;      // pad f32 rows -> conflict-free banks
  constexpr int CPT      = N / 16;     // columns per epilogue thread

  __shared__ float Cs[16][LDSS];       // 16 output rows staged per pass
  __shared__ float psum[16][16];
  __shared__ float psq[16][16];

  const int tid  = threadIdx.x;
  const int wave = tid >> 5;
  const int lane = tid & 31;
  const int wn   = wave % WAVES_N;
  const int wm   = wave / WAVES_N;
  const int rowBlock = blockIdx.x * MBLOCK;
  const int rowWave  = rowBlock + wm * MT * 16;

  // A fragment lane pattern (ISA 7.12.2, 16-bit 16x32 A):
  //   lanes 0-15 -> M=lane,   K kh=0: 0..7 & 16..23
  //   lanes16-31 -> M=lane-16,K kh=8: 8..15 & 24..31
  const int mlocal = lane & 15;
  const int khalf  = (lane >> 4) * 8;

  v8f acc[MT][NTW];
#pragma unroll
  for (int i = 0; i < MT; ++i)
#pragma unroll
    for (int j = 0; j < NTW; ++j)
#pragma unroll
      for (int e = 0; e < 8; ++e) acc[i][j][e] = 0.0f;

  // 32-bit byte-offset induction variables against SGPR base pointers.
  const char* a1b = (const char*)A1;
  const char* a2b = (const char*)A2;
  const char* wb  = (const char*)Wp;
  int offB[NTW];
#pragma unroll
  for (int j = 0; j < NTW; ++j)
    offB[j] = (((wn * NTW + j) * KT * 32 + lane) * 16) * 2;   // bytes
  int offA[MT];
#pragma unroll
  for (int i = 0; i < MT; ++i)
    offA[i] = ((rowWave + i * 16 + mlocal) * lda1 + khalf) * 2;

  int kt = 0;
  for (int seg = 0; seg < 2; ++seg) {
    const int kend = (seg == 0) ? ksplit : KT;
    if (seg == 1) {
#pragma unroll
      for (int i = 0; i < MT; ++i)
        offA[i] = ((rowWave + i * 16 + mlocal) * lda2 + khalf) * 2;
    }
    for (; kt < kend; ++kt) {
      const char* ab = (seg == 0) ? a1b : a2b;
      v16bf a[MT], b[NTW];
#pragma unroll
      for (int i = 0; i < MT; ++i) {
        v8bf lo = *(const v8bf*)(ab + offA[i]);        // K kh..kh+7
        v8bf hi = *(const v8bf*)(ab + offA[i] + 32);   // K 16+kh..16+kh+7
        a[i] = combine16(lo, hi);
        offA[i] += 64;                                  // next 32-wide k-tile
      }
#pragma unroll
      for (int j = 0; j < NTW; ++j) {
        v8bf lo = *(const v8bf*)(wb + offB[j]);
        v8bf hi = *(const v8bf*)(wb + offB[j] + 16);
        b[j] = combine16(lo, hi);
        offB[j] += 1024;                                // 32x16 bf16 fragment
      }
#pragma unroll
      for (int j = 0; j < NTW; ++j)
#pragma unroll
        for (int i = 0; i < MT; ++i)
          acc[i][j] = __builtin_amdgcn_wmma_f32_16x16x32_bf16(
              false, a[i], false, b[j], (short)0, acc[i][j], false, false);
    }
  }

  // Epilogue: one 16-row pass per m-tile of the block.
  const int row16 = tid >> 4;
  const int seg2  = tid & 15;
  const int c0    = seg2 * CPT;

  for (int mp = 0; mp < TOTAL_MT; ++mp) {
    __syncthreads();                       // previous pass fully consumed
    if (wm == mp / MT) {                   // owning waves dump C frags to LDS
      const int i  = mp % MT;
      const int rb = (lane >> 4) * 8;      // VGPR r: M=r (lo lanes) / 8+r (hi)
      const int cl = lane & 15;
#pragma unroll
      for (int j = 0; j < NTW; ++j) {
        const int colBase = (wn * NTW + j) * 16;
#pragma unroll
        for (int r = 0; r < 8; ++r)
          Cs[rb + r][colBase + cl] = acc[i][j][r];
      }
    }
    __syncthreads();

    const int gRow = rowBlock + mp * 16 + row16;
    float mean = 0.f, rstd = 1.f;
    if (LN) {
      float s = 0.f, s2 = 0.f;
      for (int c = c0; c < c0 + CPT; ++c) {
        float v = Cs[row16][c] + bias[c];
        s += v; s2 += v * v;
      }
      psum[row16][seg2] = s; psq[row16][seg2] = s2;
      __syncthreads();
      float S = 0.f, S2 = 0.f;
#pragma unroll
      for (int j = 0; j < 16; ++j) { S += psum[row16][j]; S2 += psq[row16][j]; }
      mean = S / (float)N;
      float var = S2 / (float)N - mean * mean;
      rstd = rsqrtf(var + 1e-5f);
    }
    bool doStore = true;
    if (MASKED) doStore = (stage[gRow] == stageVal);
    for (int c = c0; c < c0 + CPT; ++c) {
      float v = Cs[row16][c] + bias[c];
      if (LN)   v = (v - mean) * rstd * gamma[c] + beta[c];
      if (RELU) v = v > 0.f ? v : 0.f;
      if (doStore) {
        if (OUTF32) ((float*)outp)[(size_t)gRow * ldo + c] = v;
        else        ((bf16_t*)outp)[(size_t)gRow * ldo + c] = (bf16_t)v;
      }
    }
  }
}

// ---- weight packing: f32 row-major (K,N) -> WMMA B-fragment bf16 -----------
__global__ void pack_w(const float* __restrict__ W, bf16_t* __restrict__ dst,
                       int K, int N) {
  int p = blockIdx.x * 256 + threadIdx.x;
  if (p >= K * N) return;
  const int j    = p & 15;
  const int lane = (p >> 4) & 31;
  const int KT   = K >> 5;
  const int kt   = (p >> 9) % KT;
  const int nt   = (p >> 9) / KT;
  const int n    = nt * 16 + (lane & 15);
  const int kh   = (lane >> 4) * 8;
  const int k    = kt * 32 + ((j < 8) ? (kh + j) : (16 + kh + (j - 8)));
  dst[p] = (bf16_t)W[(size_t)k * N + n];
}

// ---- input assembly --------------------------------------------------------
__global__ void build_cs(const float* __restrict__ prob,
                         const float* __restrict__ hist,
                         bf16_t* __restrict__ cs, int B) {
  int idx = blockIdx.x * 256 + threadIdx.x;
  if (idx >= B * 256) return;
  int row = idx >> 8, c = idx & 255;
  float v = (c < 128) ? prob[row * 128 + c] : hist[row * 128 + (c - 128)];
  cs[idx] = (bf16_t)v;
}

__global__ void embed_kernel(const int* __restrict__ tables,
                             const float* __restrict__ emb,
                             bf16_t* __restrict__ tab,
                             int* __restrict__ stage, int B) {
  int idx = blockIdx.x * 256 + threadIdx.x;
  int row = idx / 96;
  int j   = idx - row * 96;
  if (row >= B) return;
  if (j == 0) {
    int cnt = 0;
#pragma unroll
    for (int s = 0; s < 4; ++s) {
      int t = tables[row * 4 + s];
      cnt += (t >= 0 && t < 32) ? 1 : 0;
    }
    stage[row] = cnt > 3 ? 3 : cnt;
  }
  int slot = j >> 5, e = j & 31;
  int t = tables[row * 4 + slot];
  float v = (t >= 0 && t < 32) ? emb[t * 32 + e] : 0.0f;  // masked / pad row
  tab[(size_t)row * 96 + j] = (bf16_t)v;
}

// Build dense padded head weights: [512x64] = card|bid|ann|v1..v4|0,
// and padded action head [512x16] = ac2|0, plus their bias vectors.
__global__ void build_heads(const float* cw, const float* cb,
                            const float* bw, const float* bb,
                            const float* aw, const float* ab,
                            const float* v1w, const float* v1b,
                            const float* v2w, const float* v2b,
                            const float* v3w, const float* v3b,
                            const float* v4w, const float* v4b,
                            const float* ac2w, const float* ac2b,
                            float* Hw, float* Hb, float* Aw, float* Ab) {
  int idx = blockIdx.x * 256 + threadIdx.x;
  if (idx < 512 * 64) {
    int k = idx >> 6, n = idx & 63;
    float v = 0.f;
    if (n < 32)      v = cw[k * 32 + n];
    else if (n < 41) v = bw[k * 9 + (n - 32)];
    else if (n < 51) v = aw[k * 10 + (n - 41)];
    else if (n < 55) { const float* vw[4] = {v1w, v2w, v3w, v4w}; v = vw[n - 51][k]; }
    Hw[idx] = v;
  } else if (idx < 512 * 64 + 512 * 16) {
    int p = idx - 512 * 64;
    int k = p >> 4, n = p & 15;
    Aw[p] = (n < 2) ? ac2w[k * 2 + n] : 0.f;
  } else if (idx < 512 * 64 + 512 * 16 + 64) {
    int n = idx - (512 * 64 + 512 * 16);
    float v = 0.f;
    if (n < 32)      v = cb[n];
    else if (n < 41) v = bb[n - 32];
    else if (n < 51) v = ab[n - 41];
    else if (n < 55) { const float* vb[4] = {v1b, v2b, v3b, v4b}; v = vb[n - 51][0]; }
    Hb[n] = v;
  } else if (idx < 512 * 64 + 512 * 16 + 64 + 16) {
    int n = idx - (512 * 64 + 512 * 16 + 64);
    Ab[n] = (n < 2) ? ac2b[n] : 0.f;
  }
}

// Final output: [B,54] = action(2) | card(32) | bid(9) | ann(10) | value(1)
__global__ void assemble(const float* __restrict__ Aout,
                         const float* __restrict__ Hout,
                         const int* __restrict__ stage,
                         float* __restrict__ out, int B) {
  int idx = blockIdx.x * 256 + threadIdx.x;
  int row = idx >> 6, c = idx & 63;
  if (row >= B || c >= 54) return;
  float v;
  if (c < 2)       v = Aout[(size_t)row * 16 + c];
  else if (c < 34) v = Hout[(size_t)row * 64 + (c - 2)];
  else if (c < 43) v = Hout[(size_t)row * 64 + 32 + (c - 34)];
  else if (c < 53) v = Hout[(size_t)row * 64 + 41 + (c - 43)];
  else             v = Hout[(size_t)row * 64 + 51 + stage[row]];
  out[(size_t)row * 54 + c] = v;
}

// ---------------------------------------------------------------------------
extern "C" void kernel_launch(void* const* d_in, const int* in_sizes, int n_in,
                              void* d_out, int out_size, void* d_ws, size_t ws_size,
                              hipStream_t stream) {
  (void)n_in; (void)out_size; (void)ws_size;
  const int B = in_sizes[0] / 128;                 // 65536
  const float* prob   = (const float*)d_in[0];
  const float* hist   = (const float*)d_in[1];
  const int*   tables = (const int*)d_in[2];
  const float* emb    = (const float*)d_in[3];
  auto W = [&](int i) { return (const float*)d_in[i]; };

  size_t off = 0;
  auto take = [&](size_t bytes) -> void* {
    void* p = (char*)d_ws + off;
    off += (bytes + 255) & ~(size_t)255;
    return p;
  };

  // packed bf16 weights (~8.7 MB -> L2-resident)
  bf16_t* ac1p = (bf16_t*)take(256 * 512 * 2);
  bf16_t* pe1p = (bf16_t*)take(256 * 512 * 2);
  bf16_t* pe2p = (bf16_t*)take(512 * 512 * 2);
  bf16_t* te1p = (bf16_t*)take(32 * 512 * 2);
  bf16_t* te2p = (bf16_t*)take(64 * 512 * 2);
  bf16_t* te3p = (bf16_t*)take(96 * 512 * 2);
  bf16_t* c1ap = (bf16_t*)take(512 * 512 * 2);
  bf16_t* c1bp = (bf16_t*)take(512 * 512 * 2);
  bf16_t* c2ap = (bf16_t*)take(1024 * 512 * 2);
  bf16_t* c2bp = (bf16_t*)take(512 * 512 * 2);
  bf16_t* c3ap = (bf16_t*)take(1024 * 512 * 2);
  bf16_t* c3bp = (bf16_t*)take(512 * 512 * 2);
  bf16_t* c3cp = (bf16_t*)take(512 * 512 * 2);
  bf16_t* c4ap = (bf16_t*)take(1024 * 512 * 2);
  bf16_t* c4bp = (bf16_t*)take(512 * 512 * 2);
  bf16_t* c4cp = (bf16_t*)take(512 * 512 * 2);
  bf16_t* c4dp = (bf16_t*)take(512 * 512 * 2);
  bf16_t* headp = (bf16_t*)take(512 * 64 * 2);
  bf16_t* ac2p  = (bf16_t*)take(512 * 16 * 2);
  float* Hw  = (float*)take(512 * 64 * 4);
  float* Awd = (float*)take(512 * 16 * 4);
  float* Hb  = (float*)take(64 * 4);
  float* Ab  = (float*)take(16 * 4);
  // activations
  bf16_t* cs    = (bf16_t*)take((size_t)B * 256 * 2);
  bf16_t* tab   = (bf16_t*)take((size_t)B * 96 * 2);
  int*    stg   = (int*)take((size_t)B * 4);
  bf16_t* hand  = (bf16_t*)take((size_t)B * 512 * 2);
  bf16_t* u     = (bf16_t*)take((size_t)B * 512 * 2);
  bf16_t* t     = (bf16_t*)take((size_t)B * 512 * 2);
  bf16_t* fused = (bf16_t*)take((size_t)B * 512 * 2);   // also used as "h"
  float*  Hout  = (float*)take((size_t)B * 64 * 4);
  float*  Aout  = (float*)take((size_t)B * 16 * 4);

  auto pack = [&](const float* src, bf16_t* dst, int K, int N) {
    pack_w<<<(K * N + 255) / 256, 256, 0, stream>>>(src, dst, K, N);
  };
  pack(W(4),  ac1p, 256, 512);  pack(W(8),  pe1p, 256, 512);
  pack(W(12), pe2p, 512, 512);  pack(W(16), te1p, 32, 512);
  pack(W(20), te2p, 64, 512);   pack(W(24), te3p, 96, 512);
  pack(W(28), c1ap, 512, 512);  pack(W(30), c1bp, 512, 512);
  pack(W(32), c2ap, 1024, 512); pack(W(34), c2bp, 512, 512);
  pack(W(36), c3ap, 1024, 512); pack(W(40), c3bp, 512, 512);
  pack(W(44), c3cp, 512, 512);  pack(W(46), c4ap, 1024, 512);
  pack(W(50), c4bp, 512, 512);  pack(W(54), c4cp, 512, 512);
  pack(W(58), c4dp, 512, 512);

  build_heads<<<(512 * 64 + 512 * 16 + 64 + 16 + 255) / 256, 256, 0, stream>>>(
      W(60), W(61), W(62), W(63), W(64), W(65),
      W(66), W(67), W(68), W(69), W(70), W(71), W(72), W(73),
      W(6), W(7), Hw, Hb, Awd, Ab);
  pack(Hw, headp, 512, 64);
  pack(Awd, ac2p, 512, 16);

  build_cs<<<(B * 256 + 255) / 256, 256, 0, stream>>>(prob, hist, cs, B);
  embed_kernel<<<(B * 96 + 255) / 256, 256, 0, stream>>>(tables, emb, tab, stg, B);

  dim3 blk(256);
  const int gM = B / 32, gH = B / 64, gA = B / 256;
  // a1 = relu(ac1(cs)) -> u
  gemm_kernel<32, 8, 2, false, true,  false, false><<<gM, blk, 0, stream>>>(
      cs, 256, cs, 256, 8, 8, ac1p, W(5), W(5), W(5), u, 512, stg, 0);
  // action = a1 @ ac2(pad16) -> Aout (f32)
  gemm_kernel<1, 1, 2, false, false, true, false><<<gA, blk, 0, stream>>>(
      u, 512, u, 512, 16, 16, ac2p, Ab, Ab, Ab, Aout, 16, stg, 0);
  // h = relu(ln1(pe1(cs))) -> fused (as scratch)
  gemm_kernel<32, 8, 2, true, true, false, false><<<gM, blk, 0, stream>>>(
      cs, 256, cs, 256, 8, 8, pe1p, W(9), W(10), W(11), fused, 512, stg, 0);
  // hand = ln2(pe2(h))
  gemm_kernel<32, 8, 2, true, false, false, false><<<gM, blk, 0, stream>>>(
      fused, 512, fused, 512, 16, 16, pe2p, W(13), W(14), W(15), hand, 512, stg, 0);
  // --- case 2: t1 -> f2 ---
  gemm_kernel<32, 8, 2, true, true, false, false><<<gM, blk, 0, stream>>>(
      tab, 96, tab, 96, 1, 1, te1p, W(17), W(18), W(19), t, 512, stg, 0);
  gemm_kernel<32, 8, 2, false, true, false, false><<<gM, blk, 0, stream>>>(
      hand, 512, t, 512, 16, 32, c2ap, W(33), W(33), W(33), u, 512, stg, 0);
  gemm_kernel<32, 8, 2, false, false, false, true><<<gM, blk, 0, stream>>>(
      u, 512, u, 512, 16, 16, c2bp, W(35), W(35), W(35), fused, 512, stg, 1);
  // --- case 3: t2 -> f3 ---
  gemm_kernel<32, 8, 2, true, true, false, false><<<gM, blk, 0, stream>>>(
      tab, 96, tab, 96, 2, 2, te2p, W(21), W(22), W(23), t, 512, stg, 0);
  gemm_kernel<32, 8, 2, true, true, false, false><<<gM, blk, 0, stream>>>(
      hand, 512, t, 512, 16, 32, c3ap, W(37), W(38), W(39), u, 512, stg, 0);
  gemm_kernel<32, 8, 2, true, true, false, false><<<gM, blk, 0, stream>>>(
      u, 512, u, 512, 16, 16, c3bp, W(41), W(42), W(43), t, 512, stg, 0);
  gemm_kernel<32, 8, 2, false, false, false, true><<<gM, blk, 0, stream>>>(
      t, 512, t, 512, 16, 16, c3cp, W(45), W(45), W(45), fused, 512, stg, 2);
  // --- case 4: t3 -> f4 ---
  gemm_kernel<32, 8, 2, true, true, false, false><<<gM, blk, 0, stream>>>(
      tab, 96, tab, 96, 3, 3, te3p, W(25), W(26), W(27), t, 512, stg, 0);
  gemm_kernel<32, 8, 2, true, true, false, false><<<gM, blk, 0, stream>>>(
      hand, 512, t, 512, 16, 32, c4ap, W(47), W(48), W(49), u, 512, stg, 0);
  gemm_kernel<32, 8, 2, true, true, false, false><<<gM, blk, 0, stream>>>(
      u, 512, u, 512, 16, 16, c4bp, W(51), W(52), W(53), t, 512, stg, 0);
  gemm_kernel<32, 8, 2, true, true, false, false><<<gM, blk, 0, stream>>>(
      t, 512, t, 512, 16, 16, c4cp, W(55), W(56), W(57), u, 512, stg, 0);
  gemm_kernel<32, 8, 2, false, false, false, true><<<gM, blk, 0, stream>>>(
      u, 512, u, 512, 16, 16, c4dp, W(59), W(59), W(59), fused, 512, stg, 3);
  // --- case 1: f1 ---
  gemm_kernel<32, 8, 2, false, true, false, false><<<gM, blk, 0, stream>>>(
      hand, 512, hand, 512, 16, 16, c1ap, W(29), W(29), W(29), u, 512, stg, 0);
  gemm_kernel<32, 8, 2, false, false, false, true><<<gM, blk, 0, stream>>>(
      u, 512, u, 512, 16, 16, c1bp, W(31), W(31), W(31), fused, 512, stg, 0);
  // heads: Hout = fused @ [card|bid|ann|v] (N=64, f32)
  gemm_kernel<4, 4, 2, false, false, true, false><<<gH, blk, 0, stream>>>(
      fused, 512, fused, 512, 16, 16, headp, Hb, Hb, Hb, Hout, 64, stg, 0);

  assemble<<<(B * 64) / 256, 256, 0, stream>>>(Aout, Hout, stg, (float*)d_out, B);
}
